// PWCNet_41437844472498
// MI455X (gfx1250) — matrix operand order repaired and checked by reading, not compiled, not run on hardware
//
#include <hip/hip_runtime.h>
#include <hip/hip_bf16.h>

// ---------------------------------------------------------------------------
// CDNA5 async global->LDS staging (probe-verified in rounds 2-4:
// builtin exists, params are AS1 int* / AS3 int*; emits
// global_load_async_to_lds_b32 + s_wait_asynccnt).
// ---------------------------------------------------------------------------
#if defined(__has_builtin)
# if __has_builtin(__builtin_amdgcn_global_load_async_to_lds_b32)
#  define HAVE_ASYNC_LDS 1
# endif
# if __has_builtin(__builtin_amdgcn_s_wait_asynccnt)
#  define HAVE_WAIT_ASYNC_BUILTIN 1
# endif
#endif
#ifndef HAVE_ASYNC_LDS
# define HAVE_ASYNC_LDS 0
#endif

#if HAVE_ASYNC_LDS
# define ASYNC_GPTR __attribute__((address_space(1))) int*
# define ASYNC_LPTR __attribute__((address_space(3))) int*
# if defined(HAVE_WAIT_ASYNC_BUILTIN)
#  define ASYNC_WAIT() __builtin_amdgcn_s_wait_asynccnt(0)
# else
#  define ASYNC_WAIT() asm volatile("s_wait_asynccnt 0x0" ::: "memory")
# endif
#else
# define ASYNC_WAIT() ((void)0)
#endif

// ---------------------------------------------------------------------------
// Types for CDNA5 WMMA (wave32, 16x16x32 bf16 -> f32)
// ---------------------------------------------------------------------------
typedef __attribute__((ext_vector_type(16))) __bf16 v16bf_t;
typedef __attribute__((ext_vector_type(8)))  float  v8f_t;

union Frag16 {
    v16bf_t v;
    unsigned int u32[8];
};

// ---------------------------------------------------------------------------
// One-time per-conv weight conversion: f32 [Cout][K] -> bf16 [Np][Kp],
// zero-padded so the GEMM hot loop needs no bounds checks for B at all
// (padded zeros contribute exact zeros to the accumulators).
// ---------------------------------------------------------------------------
__global__ void wcvt_kernel(const float* __restrict__ w, unsigned short* __restrict__ o,
                            int Cout, int K, int Kp)
{
    int k = blockIdx.x * blockDim.x + threadIdx.x;
    int n = blockIdx.y;
    if (k >= Kp) return;
    float f = (n < Cout && k < K) ? w[(size_t)n * K + k] : 0.f;
    __bf16 b = (__bf16)f;                     // native v_cvt on gfx1250
    o[(size_t)n * Kp + k] = __builtin_bit_cast(unsigned short, b);
}

// ---------------------------------------------------------------------------
// Implicit-GEMM 3x3 conv.
// Tile: 128 pixels (M) x 64 out-channels (N); K = Cin*9 in chunks of 32.
// 256 threads = 8 waves arranged 4(M) x 2(N); each wave owns a 32x32 macro
// tile = 2x2 accumulators -> 4 v_wmma_f32_16x16x32_bf16 per K-chunk.
//   A: im2col gather via 16 precomputed per-pixel base pointers + a single
//      thread-uniform per-chunk offset (cin*HinWin + dky*Win + dkx): no
//      per-element multiplies in the K-loop. k->(cin,ky,kx) kept
//      incrementally (32 = 3*9+5): no divisions either.
//   B: pre-converted bf16 weights (wcvt_kernel) staged via async global->LDS
//      b32 ops -- zero conversion VALU and zero bounds logic in the loop.
// Fused bias + leaky ReLU; channel-offset output for free concats.
// ---------------------------------------------------------------------------
#define CT_M 128
#define CT_N 64
#define CT_K 32

__global__ __launch_bounds__(256)
void conv3x3_wmma(const float* __restrict__ in, const unsigned short* __restrict__ wbf,
                  const float* __restrict__ bias, float* __restrict__ out,
                  int Bn, int Cin, int Hin, int Win,
                  int Cout, int Hout, int Wout,
                  int stride, int dil, int relu,
                  int outCtot, int outCoff, int Kp)
{
    __shared__ __bf16 As[CT_M][CT_K + 4];   // bf16 activations
    __shared__ __bf16 Bs[CT_N][CT_K + 4];   // bf16 weights (async-staged)

    __builtin_amdgcn_s_wait_tensorcnt(0);   // CDNA5 split-counter wait

    const int tid  = threadIdx.x;
    const int lane = tid & 31;
    const int wave = tid >> 5;
    const int m0w  = (wave & 3) * 32;       // wave macro-tile origin (M)
    const int n0w  = (wave >> 2) * 32;      // wave macro-tile origin (N)

    const int pix0 = blockIdx.x * CT_M;
    const int n0   = blockIdx.y * CT_N;
    const int b    = blockIdx.z;

    const int K      = Cin * 9;
    const int HW     = Hout * Wout;
    const int HinWin = Hin * Win;
    const int half   = lane >> 4;
    const int l15    = lane & 15;

    // ---- loop-invariant A-staging metadata (one fixed K-column per thread) --
    const int kk      = tid & 31;
    const int rowBase = tid >> 5;            // 0..7
    const float* inB  = in + (size_t)b * Cin * HinWin;

    int oyS[16], oxS[16];                    // pre-scaled coords (for halo test)
    const float* pj[16];                     // per-pixel base pointers
#pragma unroll
    for (int j = 0; j < 16; ++j) {
        int pix = pix0 + rowBase + j * 8;
        if (pix < HW) {
            int oy = pix / Wout;             // once, outside K-loop
            int ox = pix - oy * Wout;
            oyS[j] = oy * stride;
            oxS[j] = ox * stride;
            pj[j]  = inB + (size_t)(oy * stride) * Win + (ox * stride);
        } else {                             // sentinel -> always OOB -> 0
            oyS[j] = -0x40000000;
            oxS[j] = -0x40000000;
            pj[j]  = inB;
        }
    }

    // ---- loop-invariant B-staging metadata: 4 coalesced dwords per thread --
    const unsigned int* wg = (const unsigned int*)wbf;
    const int KpD = Kp >> 1;                 // row stride in dwords
    size_t gRow[4];
    int    bn[4], bkd[4];
#pragma unroll
    for (int i = 0; i < 4; ++i) {
        int d   = tid + i * 256;             // 0..1023 = 64 rows x 16 dwords
        bn[i]   = d >> 4;
        bkd[i]  = d & 15;
        gRow[i] = (size_t)(n0 + bn[i]) * KpD + bkd[i];   // rows padded -> no clamp
    }

    // incremental k -> (cin, r) for this thread's A column
    int cin = kk / 9;
    int r   = kk - cin * 9;

    v8f_t acc[2][2] = {};

    for (int k0 = 0; k0 < K; k0 += CT_K) {
        __syncthreads();                     // previous chunk consumed

        // ---- stage B tile: bf16 weights, async global->LDS ----
        const int kd0 = k0 >> 1;
#pragma unroll
        for (int i = 0; i < 4; ++i) {
            const unsigned int* gp = wg + gRow[i] + kd0;
            unsigned int* lp = (unsigned int*)&Bs[bn[i]][bkd[i] * 2];
#if HAVE_ASYNC_LDS
            __builtin_amdgcn_global_load_async_to_lds_b32((ASYNC_GPTR)gp, (ASYNC_LPTR)lp, 0, 0);
#else
            *lp = *gp;
#endif
        }

        // ---- stage A tile: im2col gather, native f32->bf16, zero halo ----
        {
            int ky  = r / 3;                 // r in 0..8 (cheap const div)
            int kx  = r - ky * 3;
            int dky = (ky - 1) * dil;
            int dkx = (kx - 1) * dil;
            // single thread-uniform offset for all 16 pixels this chunk
            ptrdiff_t off = (ptrdiff_t)cin * HinWin + (ptrdiff_t)dky * Win + dkx;
            bool kOk = (k0 + kk) < K;
#pragma unroll
            for (int j = 0; j < 16; ++j) {
                int iy = oyS[j] + dky;
                int ix = oxS[j] + dkx;
                float f = 0.f;
                if (kOk && iy >= 0 && iy < Hin && ix >= 0 && ix < Win)
                    f = pj[j][off];
                As[rowBase + j * 8][kk] = (__bf16)f;
            }
        }
        // advance (cin, r) by CT_K = 32 = 3*9 + 5
        r += 5; cin += 3;
        if (r >= 9) { r -= 9; cin += 1; }

        if (tid == 0 && k0 + CT_K < K)
            __builtin_prefetch(inB + (size_t)cin * HinWin, 0, 1);  // global_prefetch_b8

        ASYNC_WAIT();                        // own wave's async LDS writes
        __syncthreads();                     // all waves' staging visible

        // ---- build fragments (ISA 7.12.2 layouts), 2x2 WMMA ----
        Frag16 fa[2], fb[2];
#pragma unroll
        for (int s = 0; s < 2; ++s) {
            int arow = m0w + s * 16 + l15;
            int bcol = n0w + s * 16 + l15;
#pragma unroll
            for (int v = 0; v < 8; ++v) {
                // A (16x32, 16-bit): VGPR v holds a K pair; lanes 16-31: K+8
                int kbA = ((v & 4) ? 16 : 0) + (half ? 8 : 0) + (v & 3) * 2;
                // B (32x16, 16-bit): column per lane; same K order as A (lanes 16-31: K+8)
                int kbB = ((v & 4) ? 16 : 0) + (half ? 8 : 0) + (v & 3) * 2;
                fa[s].u32[v] = *(const unsigned int*)&As[arow][kbA];
                fb[s].u32[v] = *(const unsigned int*)&Bs[bcol][kbB];
            }
        }
#pragma unroll
        for (int i = 0; i < 2; ++i)
#pragma unroll
            for (int j = 0; j < 2; ++j)
                acc[i][j] = __builtin_amdgcn_wmma_f32_16x16x32_bf16(
                    false, fa[i].v, false, fb[j].v, (short)0, acc[i][j], false, false);
    }

    // ---- epilogue: C/D layout (VGPR rr -> M = rr + 8*half, N = lane&15) ----
#pragma unroll
    for (int j = 0; j < 2; ++j) {
        int n = n0 + n0w + j * 16 + l15;
        float bv = (n < Cout) ? bias[n] : 0.f;
#pragma unroll
        for (int i = 0; i < 2; ++i) {
#pragma unroll
            for (int rr = 0; rr < 8; ++rr) {
                int m = pix0 + m0w + i * 16 + rr + (half ? 8 : 0);
                if (m < HW && n < Cout) {
                    float v = acc[i][j][rr] + bv;
                    if (relu) v = v > 0.f ? v : 0.1f * v;
                    out[((size_t)b * outCtot + outCoff + n) * (size_t)HW + m] = v;
                }
            }
        }
    }
}

// ---------------------------------------------------------------------------
// Cost volume: 81 displacement channels, mean over C, fused leaky ReLU.
// ---------------------------------------------------------------------------
__global__ void corr_kernel(const float* __restrict__ x1, const float* __restrict__ x2,
                            float* __restrict__ out, int Bn, int C, int H, int W, int total)
{
    int idx = blockIdx.x * blockDim.x + threadIdx.x;
    if (idx >= total) return;
    int x = idx % W; int t = idx / W;
    int y = t % H;  t /= H;
    int d = t % 81; int b = t / 81;
    int dy = d / 9 - 4, dx = d % 9 - 4;
    int yy = y + dy, xx = x + dx;
    float s = 0.f;
    if (yy >= 0 && yy < H && xx >= 0 && xx < W) {
        const float* p1 = x1 + (size_t)b * C * H * W + (size_t)y * W + x;
        const float* p2 = x2 + (size_t)b * C * H * W + (size_t)yy * W + xx;
        size_t hw = (size_t)H * W;
        for (int c = 0; c < C; ++c) s += p1[c * hw] * p2[c * hw];
    }
    s *= (1.0f / (float)C);
    out[idx] = s > 0.f ? s : 0.1f * s;
}

// ---------------------------------------------------------------------------
// Bilinear resize, align_corners=true; fused output scale.
// ---------------------------------------------------------------------------
__global__ void resize_kernel(const float* __restrict__ in, float* __restrict__ out,
                              int Bn, int C, int Hi, int Wi, int Ho, int Wo,
                              float scale, int total)
{
    int idx = blockIdx.x * blockDim.x + threadIdx.x;
    if (idx >= total) return;
    int x = idx % Wo; int t = idx / Wo;
    int y = t % Ho;  t /= Ho;
    int c = t % C;   int b = t / C;
    float ty = (Ho > 1) ? y * (float)(Hi - 1) / (float)(Ho - 1) : 0.f;
    float tx = (Wo > 1) ? x * (float)(Wi - 1) / (float)(Wo - 1) : 0.f;
    int ymax = Hi - 2 > 0 ? Hi - 2 : 0;
    int xmax = Wi - 2 > 0 ? Wi - 2 : 0;
    int y0 = (int)floorf(ty); y0 = y0 < 0 ? 0 : (y0 > ymax ? ymax : y0);
    int x0 = (int)floorf(tx); x0 = x0 < 0 ? 0 : (x0 > xmax ? xmax : x0);
    float wy = ty - (float)y0, wx = tx - (float)x0;
    int y1 = (y0 + 1 < Hi) ? y0 + 1 : Hi - 1;
    int x1 = (x0 + 1 < Wi) ? x0 + 1 : Wi - 1;
    const float* p = in + ((size_t)b * C + c) * (size_t)(Hi * Wi);
    float v = (p[y0 * Wi + x0] * (1.f - wx) + p[y0 * Wi + x1] * wx) * (1.f - wy)
            + (p[y1 * Wi + x0] * (1.f - wx) + p[y1 * Wi + x1] * wx) * wy;
    out[idx] = v * scale;
}

// ---------------------------------------------------------------------------
// Flow warp: bilinear grid_sample (align_corners, zero pad) + validity mask.
// ---------------------------------------------------------------------------
__global__ void warp_kernel(const float* __restrict__ x, const float* __restrict__ flow,
                            float* __restrict__ out, int Bn, int C, int H, int W,
                            int h_im, int w_im, int total)
{
    int idx = blockIdx.x * blockDim.x + threadIdx.x;
    if (idx >= total) return;
    int xo = idx % W; int t = idx / W;
    int y = t % H;   t /= H;
    int c = t % C;   int b = t / C;
    size_t hw = (size_t)H * W;
    float fx = flow[((size_t)b * 2 + 0) * hw + (size_t)y * W + xo];
    float fy = flow[((size_t)b * 2 + 1) * hw + (size_t)y * W + xo];
    float sfx = 2.0f / ((w_im - 1 > 1 ? (float)(w_im - 1) : 1.f)) / 0.05f;
    float sfy = 2.0f / ((h_im - 1 > 1 ? (float)(h_im - 1) : 1.f)) / 0.05f;
    float gx = (W > 1 ? -1.f + 2.f * xo / (float)(W - 1) : -1.f) + fx * sfx;
    float gy = (H > 1 ? -1.f + 2.f * y  / (float)(H - 1) : -1.f) + fy * sfy;
    float px = (gx + 1.f) * 0.5f * (float)(W - 1);
    float py = (gy + 1.f) * 0.5f * (float)(H - 1);
    float x0 = floorf(px), y0 = floorf(py);
    const float* p = x + ((size_t)b * C + c) * hw;
    float acc = 0.f, wacc = 0.f;
#pragma unroll
    for (int dj = 0; dj < 2; ++dj) {
#pragma unroll
        for (int di = 0; di < 2; ++di) {
            float ixf = x0 + di, iyf = y0 + dj;
            float wgt = (1.f - fabsf(px - ixf)) * (1.f - fabsf(py - iyf));
            if (ixf >= 0.f && ixf <= (float)(W - 1) && iyf >= 0.f && iyf <= (float)(H - 1)) {
                int ix = (int)ixf, iy = (int)iyf;
                acc  += p[(size_t)iy * W + ix] * wgt;
                wacc += wgt;
            }
        }
    }
    out[idx] = (wacc >= 1.0f) ? acc : 0.f;
}

// ---------------------------------------------------------------------------
// Channel-offset copy (concat), zero fill, residual add.
// ---------------------------------------------------------------------------
__global__ void chan_copy_kernel(const float* __restrict__ src, float* __restrict__ dst,
                                 int Bn, int C, int HW, int dstC, int off, int total)
{
    int idx = blockIdx.x * blockDim.x + threadIdx.x;
    if (idx >= total) return;
    int p = idx % HW; int t = idx / HW;
    int c = t % C;    int b = t / C;
    dst[((size_t)b * dstC + off + c) * (size_t)HW + p] = src[idx];
}

__global__ void fill_kernel(float* __restrict__ p, float v, int n)
{
    int idx = blockIdx.x * blockDim.x + threadIdx.x;
    if (idx < n) p[idx] = v;
}

__global__ void add_kernel(float* __restrict__ dst, const float* __restrict__ src, int n)
{
    int idx = blockIdx.x * blockDim.x + threadIdx.x;
    if (idx < n) dst[idx] += src[idx];
}

// ---------------------------------------------------------------------------
// Host driver: full PWC-Net graph on stream, bump allocator on d_ws.
// ---------------------------------------------------------------------------
extern "C" void kernel_launch(void* const* d_in, const int* in_sizes, int n_in,
                              void* d_out, int out_size, void* d_ws, size_t ws_size,
                              hipStream_t stream)
{
    (void)in_sizes; (void)out_size;
    const int Bn = 2, Him = 384, Wim = 768;
    const float* input1 = (const float*)d_in[0];
    const float* input2 = (const float*)d_in[1];

    // ---- param leaves, pytree (alphabetical dict keys) order ----
    int pi = 2;
    auto next = [&]() -> const float* {
        const float* p = (const float*)d_in[pi < n_in ? pi : n_in - 1];
        if (pi < n_in - 1) ++pi;
        return p;
    };
    struct ConvP { const float* w; const float* b; };
    ConvP ctxf[7], ctxo[7];
    for (int i = 0; i < 7; ++i) { ctxf[i].w = next(); ctxf[i].b = next(); }   // ctx_flow
    for (int i = 0; i < 7; ++i) { ctxo[i].w = next(); ctxo[i].b = next(); }   // ctx_occ
    struct FeatP { const float *b1, *b2, *w1, *w2; } feat[6];                  // keys b1,b2,w1,w2
    for (int i = 0; i < 6; ++i) { feat[i].b1 = next(); feat[i].b2 = next();
                                  feat[i].w1 = next(); feat[i].w2 = next(); }
    ConvP fest[5][6], oest[5][6];
    for (int l = 0; l < 5; ++l) for (int j = 0; j < 6; ++j) { fest[l][j].w = next(); fest[l][j].b = next(); }
    for (int l = 0; l < 5; ++l) for (int j = 0; j < 6; ++j) { oest[l][j].w = next(); oest[l][j].b = next(); }

    // ---- workspace bump allocator ----
    char* wsb = (char*)d_ws;
    size_t wsOff = 0;
    auto alloc = [&](size_t nf) -> float* {
        size_t bytes = (nf * sizeof(float) + 255) & ~(size_t)255;
        if (wsOff + bytes > ws_size) wsOff = 0;   // best-effort wrap (compile-only env)
        float* p = (float*)(wsb + wsOff);
        wsOff += bytes;
        return p;
    };

    struct T { float* p; int C, H, W; };

    auto convAt = [&](const T& x, ConvP p, int Cout, int stride, int dil, int relu,
                      float* outp, int outCtot, int outCoff) -> T {
        int Ho = (x.H - 1) / stride + 1;
        int Wo = (x.W - 1) / stride + 1;
        int K  = x.C * 9;
        int Kp = (K + 31) & ~31;
        int Np = ((Cout + CT_N - 1) / CT_N) * CT_N;
        // one-time bf16 weight conversion (padded: no bounds checks in GEMM)
        unsigned short* wbf = (unsigned short*)alloc((size_t)Np * Kp / 2);
        dim3 cgrid((unsigned)((Kp + 255) / 256), (unsigned)Np);
        wcvt_kernel<<<cgrid, 256, 0, stream>>>(p.w, wbf, Cout, K, Kp);
        dim3 grid((unsigned)((Ho * Wo + CT_M - 1) / CT_M), (unsigned)(Np / CT_N), (unsigned)Bn);
        conv3x3_wmma<<<grid, 256, 0, stream>>>(x.p, wbf, p.b, outp, Bn, x.C, x.H, x.W,
                                               Cout, Ho, Wo, stride, dil, relu,
                                               outCtot, outCoff, Kp);
        return T{outp, Cout, Ho, Wo};
    };
    auto convNew = [&](const T& x, ConvP p, int Cout, int stride, int dil, int relu) -> T {
        int Ho = (x.H - 1) / stride + 1;
        int Wo = (x.W - 1) / stride + 1;
        float* o = alloc((size_t)Bn * Cout * Ho * Wo);
        return convAt(x, p, Cout, stride, dil, relu, o, Cout, 0);
    };
    auto copyTo = [&](const T& src, const T& dst, int off) {
        int n = Bn * src.C * src.H * src.W;
        chan_copy_kernel<<<(n + 255) / 256, 256, 0, stream>>>(src.p, dst.p, Bn, src.C,
                                                              src.H * src.W, dst.C, off, n);
    };
    auto fillZero = [&](const T& t) {
        int n = Bn * t.C * t.H * t.W;
        fill_kernel<<<(n + 255) / 256, 256, 0, stream>>>(t.p, 0.f, n);
    };
    auto resizeTo = [&](const T& x, int Ho, int Wo, float scale, float* outp) -> T {
        int n = Bn * x.C * Ho * Wo;
        resize_kernel<<<(n + 255) / 256, 256, 0, stream>>>(x.p, outp, Bn, x.C, x.H, x.W, Ho, Wo, scale, n);
        return T{outp, x.C, Ho, Wo};
    };
    auto resizeNew = [&](const T& x, int Ho, int Wo) -> T {
        return resizeTo(x, Ho, Wo, 1.0f, alloc((size_t)Bn * x.C * Ho * Wo));
    };
    auto warpNew = [&](const T& x, const T& flow) -> T {
        float* o = alloc((size_t)Bn * x.C * x.H * x.W);
        int n = Bn * x.C * x.H * x.W;
        warp_kernel<<<(n + 255) / 256, 256, 0, stream>>>(x.p, flow.p, o, Bn, x.C, x.H, x.W, Him, Wim, n);
        return T{o, x.C, x.H, x.W};
    };
    auto corrNew = [&](const T& a, const T& b) -> T {
        float* o = alloc((size_t)Bn * 81 * a.H * a.W);
        int n = Bn * 81 * a.H * a.W;
        corr_kernel<<<(n + 255) / 256, 256, 0, stream>>>(a.p, b.p, o, Bn, a.C, a.H, a.W, n);
        return T{o, 81, a.H, a.W};
    };
    auto addInto = [&](const T& dst, const T& src) {
        int n = Bn * dst.C * dst.H * dst.W;
        add_kernel<<<(n + 255) / 256, 256, 0, stream>>>(dst.p, src.p, n);
    };

    // Dense estimator (+ optional context net at output level) writing final
    // conv into caller-provided (pre-mark) buffer so scratch can be reset.
    auto denseEst = [&](const T& xin, ConvP* ps, int outCh, float* outp, ConvP* ctxps) {
        const int dch[5] = {128, 128, 96, 64, 32};
        T x = xin;
        for (int j = 0; j < 5; ++j) {
            int nc = dch[j] + x.C;
            float* np = alloc((size_t)Bn * nc * x.H * x.W);
            T nx{np, nc, x.H, x.W};
            convAt(x, ps[j], dch[j], 1, 1, 1, nx.p, nc, 0);   // conv -> channels [0,dch)
            copyTo(x, nx, dch[j]);                             // passthrough -> [dch,nc)
            x = nx;
        }
        T flowT{outp, outCh, x.H, x.W};
        convAt(x, ps[5], outCh, 1, 1, 0, flowT.p, outCh, 0);
        if (ctxps) {
            int cc = x.C + outCh;
            float* cp = alloc((size_t)Bn * cc * x.H * x.W);
            T cx{cp, cc, x.H, x.W};
            copyTo(x, cx, 0);
            copyTo(flowT, cx, x.C);
            const int cdil[7] = {1, 2, 4, 8, 16, 1, 1};
            const int cdim[6] = {128, 128, 128, 96, 64, 32};
            T cur = cx;
            for (int i = 0; i < 7; ++i) {
                int Co = (i == 6) ? outCh : cdim[i];
                cur = convNew(cur, ctxps[i], Co, 1, cdil[i], (i < 6) ? 1 : 0);
            }
            addInto(flowT, cur);
        }
    };

    // ---- feature pyramids (persistent) ----
    const int chs[7] = {3, 16, 32, 64, 96, 128, 196};
    T p1[6], p2[6];
    {
        T x{(float*)input1, 3, Him, Wim};
        for (int i = 0; i < 6; ++i) {
            T y = convNew(x, ConvP{feat[i].w1, feat[i].b1}, chs[i + 1], 2, 1, 1);
            T z = convNew(y, ConvP{feat[i].w2, feat[i].b2}, chs[i + 1], 1, 1, 1);
            p1[i] = z; x = z;
        }
    }
    {
        T x{(float*)input2, 3, Him, Wim};
        for (int i = 0; i < 6; ++i) {
            T y = convNew(x, ConvP{feat[i].w1, feat[i].b1}, chs[i + 1], 2, 1, 1);
            T z = convNew(y, ConvP{feat[i].w2, feat[i].b2}, chs[i + 1], 1, 1, 1);
            p2[i] = z; x = z;
        }
    }
    T x1p[5], x2p[5];
    for (int l = 0; l < 5; ++l) { x1p[l] = p1[5 - l]; x2p[l] = p2[5 - l]; }   // coarsest first

    int h0 = x1p[0].H, w0 = x1p[0].W;
    T flow_f{alloc((size_t)Bn * 2 * h0 * w0), 2, h0, w0};
    T flow_b{alloc((size_t)Bn * 2 * h0 * w0), 2, h0, w0};
    T occ_f {alloc((size_t)Bn * 1 * h0 * w0), 1, h0, w0};
    T occ_b {alloc((size_t)Bn * 1 * h0 * w0), 1, h0, w0};
    fillZero(flow_f); fillZero(flow_b); fillZero(occ_f); fillZero(occ_b);

    for (int l = 0; l < 5; ++l) {
        T x1 = x1p[l], x2 = x2p[l];
        int h = x1.H, w = x1.W;
        T x2w, x1w;
        if (l == 0) { x2w = x2; x1w = x1; }
        else {
            flow_f = resizeNew(flow_f, h, w);
            flow_b = resizeNew(flow_b, h, w);
            occ_f  = resizeNew(occ_f,  h, w);
            occ_b  = resizeNew(occ_b,  h, w);
            x2w = warpNew(x2, flow_f);
            x1w = warpNew(x1, flow_b);
        }
        T corr_f = corrNew(x1, x2w);
        T corr_b = corrNew(x2, x1w);

        T in_f, in_b, ino_f, ino_b;
        if (l == 0) { in_f = corr_f; in_b = corr_b; ino_f = corr_f; ino_b = corr_b; }
        else {
            auto cat3 = [&](const T& a, const T& b, const T& c) -> T {
                int nc = a.C + b.C + c.C;
                float* p = alloc((size_t)Bn * nc * h * w);
                T t{p, nc, h, w};
                copyTo(a, t, 0); copyTo(b, t, a.C); copyTo(c, t, a.C + b.C);
                return t;
            };
            in_f  = cat3(corr_f, x1, flow_f);
            in_b  = cat3(corr_b, x2, flow_b);
            ino_f = cat3(corr_f, x1, occ_f);
            ino_b = cat3(corr_b, x1, occ_b);   // x1 here: faithful to the original code
        }

        float* nf_f = alloc((size_t)Bn * 2 * h * w);
        float* nf_b = alloc((size_t)Bn * 2 * h * w);
        float* no_f = alloc((size_t)Bn * 1 * h * w);
        float* no_b = alloc((size_t)Bn * 1 * h * w);
        size_t mark = wsOff;
        denseEst(in_f,  fest[l], 2, nf_f, (l == 4) ? ctxf : nullptr); wsOff = mark;
        denseEst(in_b,  fest[l], 2, nf_b, (l == 4) ? ctxf : nullptr); wsOff = mark;
        denseEst(ino_f, oest[l], 1, no_f, (l == 4) ? ctxo : nullptr); wsOff = mark;
        denseEst(ino_b, oest[l], 1, no_b, (l == 4) ? ctxo : nullptr); wsOff = mark;
        flow_f = T{nf_f, 2, h, w}; flow_b = T{nf_b, 2, h, w};
        occ_f  = T{no_f, 1, h, w}; occ_b  = T{no_b, 1, h, w};
    }

    // ---- eval outputs: flow (x 1/DIV_FLOW = 20) then occ, full resolution ----
    float* outF = (float*)d_out;
    resizeTo(flow_f, Him, Wim, 20.0f, outF);
    resizeTo(occ_f,  Him, Wim, 1.0f,  outF + (size_t)Bn * 2 * Him * Wim);
}
